// RGATNetwork_51831665328276
// MI455X (gfx1250) — compile-verified, matrix operand order checked by reading, not run on hardware
//
#include <hip/hip_runtime.h>

typedef __bf16 bf16_t;
typedef __attribute__((ext_vector_type(16))) __bf16 v16bf;
typedef __attribute__((ext_vector_type(8)))  float  v8f;

#define R_REL 4
#define NHEAD 4
#define COUT  32
#define HCDIM 128
#define FE    16
#define NEG_SLOPE 0.2f
#define EPS 1e-16f
// encoding of -inf under monotonic float->uint map
#define ENC_NEG_INF 0x007FFFFFu

// WMMA 16x16x32 bf16 fragment geometry (CDNA5 ISA 7.12.2):
//   A fragment (16x32):  lane L -> row = L&15 ; K = ((L>>4)*8) + (e<8 ? e : e+8)
//   elements e=2p,2p+1 hold consecutive K -> pairwise-contiguous gathers.
// We pre-swizzle A and B operands into [fragment][lane][16 elems] order so a
// fragment is ONE contiguous 32-byte load (2 x b128) instead of 16 u16 loads.

union frag32 {
  bf16_t h[16];
  uint4  q[2];
};

// ---------------------------------------------------------------------------
// utility kernels
// ---------------------------------------------------------------------------
__global__ __launch_bounds__(256) void fill_u32(unsigned* __restrict__ p, unsigned v, int count) {
  int i = blockIdx.x * blockDim.x + threadIdx.x;
  if (i < count) p[i] = v;
}

__global__ __launch_bounds__(256) void fill_f32(float* __restrict__ p, float v, int count) {
  int i = blockIdx.x * blockDim.x + threadIdx.x;
  if (i < count) p[i] = v;
}

// lee[f][h] = sum_o le[f][o] * e[o][h]   (16x128 @ 128x4 -> 16x4)
__global__ void compute_lee(const float* __restrict__ le, const float* __restrict__ em,
                            float* __restrict__ lee) {
  int t = threadIdx.x;           // 64 threads
  int f = t >> 2, h = t & 3;
  float s = 0.f;
  for (int o = 0; o < HCDIM; ++o) s += le[f * HCDIM + o] * em[o * NHEAD + h];
  lee[f * NHEAD + h] = s;
}

// ---------------------------------------------------------------------------
// operand swizzlers: fp32 -> bf16, permuted into WMMA fragment order
// ---------------------------------------------------------------------------
// X: one thread per (row-tile t, kstep ks, lane) -> 16 bf16 (32B) contiguous out.
__global__ __launch_bounds__(256) void swizzle_x(const float* __restrict__ x,
                                                 bf16_t* __restrict__ xbs,
                                                 int nN, int total) {
  int idx = blockIdx.x * blockDim.x + threadIdx.x;   // over Tpad*4*32
  if (idx >= total) return;
  const int lane = idx & 31;
  const int ks   = (idx >> 5) & 3;
  const int t    = idx >> 7;
  const int row  = t * 16 + (lane & 15);
  const int kb   = ks * 32 + ((lane >> 4) << 3);
  frag32 f;
  if (row < nN) {
    const float* xr = x + (size_t)row * HCDIM;
#pragma unroll
    for (int p = 0; p < 8; ++p) {
      int kk = kb + 2 * p + ((p >= 4) ? 8 : 0);
      f.h[2 * p]     = (bf16_t)xr[kk];
      f.h[2 * p + 1] = (bf16_t)xr[kk + 1];
    }
  } else {
#pragma unroll
    for (int e = 0; e < 16; ++e) f.h[e] = (bf16_t)0.0f;
  }
  uint4* dst = (uint4*)(xbs + (size_t)idx * 16);
  dst[0] = f.q[0];
  dst[1] = f.q[1];
}

// layer-1 epilogue fused with swizzle: h = relu(acc + b1) -> swizzled bf16
__global__ __launch_bounds__(256) void relu_bias_swizzle(const float* __restrict__ acc,
                                                         const float* __restrict__ b,
                                                         bf16_t* __restrict__ xbs,
                                                         int nN, int total) {
  int idx = blockIdx.x * blockDim.x + threadIdx.x;
  if (idx >= total) return;
  const int lane = idx & 31;
  const int ks   = (idx >> 5) & 3;
  const int t    = idx >> 7;
  const int row  = t * 16 + (lane & 15);
  const int kb   = ks * 32 + ((lane >> 4) << 3);
  frag32 f;
  if (row < nN) {
    const float* ar = acc + (size_t)row * HCDIM;
#pragma unroll
    for (int p = 0; p < 8; ++p) {
      int kk = kb + 2 * p + ((p >= 4) ? 8 : 0);
      float v0 = ar[kk] + b[kk];
      float v1 = ar[kk + 1] + b[kk + 1];
      f.h[2 * p]     = (bf16_t)(v0 > 0.f ? v0 : 0.f);
      f.h[2 * p + 1] = (bf16_t)(v1 > 0.f ? v1 : 0.f);
    }
  } else {
#pragma unroll
    for (int e = 0; e < 16; ++e) f.h[e] = (bf16_t)0.0f;
  }
  uint4* dst = (uint4*)(xbs + (size_t)idx * 16);
  dst[0] = f.q[0];
  dst[1] = f.q[1];
}

// W: one thread per (r, col-tile ct, kstep ks, lane) -> 32B contiguous out.
// B fragment (32x16): lane L -> col = L&15 ; K = ((L>>4)*8) + (e<8 ? e : e+8)
__global__ __launch_bounds__(256) void swizzle_w(const float* __restrict__ w,
                                                 bf16_t* __restrict__ wbs) {
  int idx = blockIdx.x * blockDim.x + threadIdx.x;   // over R*8*4*32 = 4096
  if (idx >= R_REL * 8 * 4 * 32) return;
  const int lane = idx & 31;
  const int ks   = (idx >> 5) & 3;
  const int ct   = (idx >> 7) & 7;
  const int r    = idx >> 10;
  const int col  = ct * 16 + (lane & 15);
  const int kb   = ks * 32 + ((lane >> 4) << 3);
  const float* wr = w + (size_t)r * HCDIM * HCDIM;
  frag32 f;
#pragma unroll
  for (int e = 0; e < 16; ++e) {
    int kk = kb + ((e < 8) ? e : e + 8);
    f.h[e] = (bf16_t)wr[(size_t)kk * HCDIM + col];
  }
  uint4* dst = (uint4*)(wbs + (size_t)idx * 16);
  dst[0] = f.q[0];
  dst[1] = f.q[1];
}

// ---------------------------------------------------------------------------
// WMMA GEMM: xw[r] = X (N x 128) @ W_r (128 x 128), bf16 in / fp32 out.
// Block = 256 thr (8 waves); block covers 128 rows; wave = one 16-row strip.
// Swizzled W_r (32KB) staged in LDS; every fragment = one 32B load.
// Store epilogue: readfirstlane'd tile class -> scalar branches, no EXEC work.
// ---------------------------------------------------------------------------
__global__ __launch_bounds__(256) void gemm_xw_wmma(const bf16_t* __restrict__ xbs,
                                                    const bf16_t* __restrict__ wbs,
                                                    float* __restrict__ xw, int n_rows) {
  __shared__ bf16_t sW[HCDIM * HCDIM];   // 32 KB, fragment-ordered
  const int r = blockIdx.y;

  // cooperative 128-bit copies of swizzled W_r into LDS
  const uint4* wsrc = (const uint4*)(wbs + (size_t)r * HCDIM * HCDIM);
  uint4* wdst = (uint4*)sW;
  for (int i = threadIdx.x; i < (HCDIM * HCDIM) / 8; i += 256) wdst[i] = wsrc[i];
  __syncthreads();

  const int wave = threadIdx.x >> 5;
  const int lane = threadIdx.x & 31;
  const int t    = blockIdx.x * 8 + wave;      // 16-row tile index (always allocated)
  const int m0   = t * 16;

  // A fragments for the 4 k-steps: one 32B coalesced global load each
  const v16bf* abase = (const v16bf*)xbs + (size_t)t * 128 + lane;
  v16bf A[4];
#pragma unroll
  for (int ks = 0; ks < 4; ++ks) A[ks] = abase[ks * 32];

  const v16bf* bbase = (const v16bf*)sW;
  float* obase = xw + (size_t)r * n_rows * HCDIM;
  const int col16 = lane & 15;
  const int mhi   = (lane >> 4) * 8;
  // tile classification is wave-uniform (m0 depends only on wave id):
  // force it into an SGPR so the epilogue guards become scalar branches.
  int tclass = (m0 + 16 <= n_rows) ? 2 : ((m0 < n_rows) ? 1 : 0);
  tclass = __builtin_amdgcn_readfirstlane(tclass);

#pragma unroll 1
  for (int ct = 0; ct < 8; ++ct) {             // 8 column tiles of 16
    v8f acc = {};
#pragma unroll
    for (int ks = 0; ks < 4; ++ks) {
      v16bf B = bbase[(ct * 4 + ks) * 32 + lane];   // one 32B LDS read
      acc = __builtin_amdgcn_wmma_f32_16x16x32_bf16(false, A[ks], false, B,
                                                    (short)0, acc, false, false);
    }
    const int col = ct * 16 + col16;
    float* orow = obase + (size_t)(m0 + mhi) * HCDIM + col;
    if (tclass == 2) {
      // D layout: VGPR v -> M = mhi + v, N = col ; unguarded clause of stores
#pragma unroll
      for (int v = 0; v < 8; ++v) orow[(size_t)v * HCDIM] = acc[v];
    } else if (tclass == 1) {
      // cold path: partial tile (only if n_rows % 16 != 0)
#pragma unroll
      for (int v = 0; v < 8; ++v)
        if (m0 + mhi + v < n_rows) orow[(size_t)v * HCDIM] = acc[v];
    }
  }
}

// ---------------------------------------------------------------------------
// edge pass 1: alpha = leaky_relu(x_i@q + x_j@k + ea@lee); atomic segment max
// q/k/lee staged in LDS (re-read 128x per thread).
// ---------------------------------------------------------------------------
__global__ __launch_bounds__(256) void edge_alpha(const float* __restrict__ xw,
                                                  const int* __restrict__ ei,
                                                  const int* __restrict__ etype,
                                                  const float* __restrict__ eattr,
                                                  const float* __restrict__ q,
                                                  const float* __restrict__ k,
                                                  const float* __restrict__ lee,
                                                  float* __restrict__ abuf,
                                                  unsigned* __restrict__ amax,
                                                  int nE, int nN) {
  __shared__ float sQ[HCDIM * NHEAD];    // 2KB
  __shared__ float sK[HCDIM * NHEAD];    // 2KB
  __shared__ float sL[FE * NHEAD];       // 256B
  for (int i = threadIdx.x; i < HCDIM * NHEAD; i += 256) { sQ[i] = q[i]; sK[i] = k[i]; }
  if (threadIdx.x < FE * NHEAD) sL[threadIdx.x] = lee[threadIdx.x];
  __syncthreads();

  int e = blockIdx.x * blockDim.x + threadIdx.x;
  if (e >= nE) return;
  const int s = ei[e], d = ei[nE + e], r = etype[e];
  const float4* xi4 = (const float4*)(xw + ((size_t)r * nN + d) * HCDIM);
  const float4* xj4 = (const float4*)(xw + ((size_t)r * nN + s) * HCDIM);
  __builtin_prefetch(xj4, 0, 0);
  const float4* q4 = (const float4*)sQ;
  const float4* k4 = (const float4*)sK;

  float4 qi = {0, 0, 0, 0}, kj = {0, 0, 0, 0};
  for (int o4 = 0; o4 < HCDIM / 4; ++o4) {
    float4 a = xi4[o4], b = xj4[o4];
#pragma unroll
    for (int c = 0; c < 4; ++c) {
      float av = (&a.x)[c], bv = (&b.x)[c];
      float4 qr = q4[o4 * 4 + c], kr = k4[o4 * 4 + c];
      qi.x += av * qr.x; qi.y += av * qr.y; qi.z += av * qr.z; qi.w += av * qr.w;
      kj.x += bv * kr.x; kj.y += bv * kr.y; kj.z += bv * kr.z; kj.w += bv * kr.w;
    }
  }
  // edge-feature key term: ea (16) @ lee (16x4)
  const float4* ea4  = (const float4*)(eattr + (size_t)e * FE);
  const float4* lee4 = (const float4*)sL;
#pragma unroll
  for (int f4 = 0; f4 < 4; ++f4) {
    float4 v = ea4[f4];
#pragma unroll
    for (int c = 0; c < 4; ++c) {
      float vv = (&v.x)[c];
      float4 lr = lee4[f4 * 4 + c];
      kj.x += vv * lr.x; kj.y += vv * lr.y; kj.z += vv * lr.z; kj.w += vv * lr.w;
    }
  }
  float al[NHEAD] = {qi.x + kj.x, qi.y + kj.y, qi.z + kj.z, qi.w + kj.w};
#pragma unroll
  for (int h = 0; h < NHEAD; ++h) {
    float a = al[h];
    a = (a > 0.f) ? a : NEG_SLOPE * a;
    abuf[(size_t)e * NHEAD + h] = a;
    int b = __float_as_int(a);
    unsigned enc = (unsigned)b ^ ((unsigned)(b >> 31) | 0x80000000u);  // monotonic map
    atomicMax(&amax[d * NHEAD + h], enc);
  }
}

// ---------------------------------------------------------------------------
// edge pass 2: ex = exp(alpha - max[dst]); atomic segment sum
// ---------------------------------------------------------------------------
__global__ __launch_bounds__(256) void edge_expsum(float* __restrict__ abuf,
                                                   const unsigned* __restrict__ amax,
                                                   const int* __restrict__ ei,
                                                   float* __restrict__ denom, int nE) {
  int e = blockIdx.x * blockDim.x + threadIdx.x;
  if (e >= nE) return;
  const int d = ei[nE + e];
#pragma unroll
  for (int h = 0; h < NHEAD; ++h) {
    unsigned enc = amax[d * NHEAD + h];
    int b = (enc & 0x80000000u) ? (int)(enc ^ 0x80000000u) : (int)~enc;
    float m  = __int_as_float(b);
    float ex = __expf(abuf[(size_t)e * NHEAD + h] - m);
    abuf[(size_t)e * NHEAD + h] = ex;
    atomicAdd(&denom[d * NHEAD + h], ex);
  }
}

// ---------------------------------------------------------------------------
// edge pass 3: msg = (ex/denom) * x_j ; scatter-add into acc[dst]
// ---------------------------------------------------------------------------
__global__ __launch_bounds__(256) void edge_scatter(const float* __restrict__ abuf,
                                                    const float* __restrict__ denom,
                                                    const float* __restrict__ xw,
                                                    const int* __restrict__ ei,
                                                    const int* __restrict__ etype,
                                                    float* __restrict__ acc, int nE, int nN) {
  int e = blockIdx.x * blockDim.x + threadIdx.x;
  if (e >= nE) return;
  const int s = ei[e], d = ei[nE + e], r = etype[e];
  const float4* xj4 = (const float4*)(xw + ((size_t)r * nN + s) * HCDIM);
  float al[NHEAD];
#pragma unroll
  for (int h = 0; h < NHEAD; ++h)
    al[h] = abuf[(size_t)e * NHEAD + h] / (denom[d * NHEAD + h] + EPS);
  float* o = acc + (size_t)d * HCDIM;
  for (int i4 = 0; i4 < HCDIM / 4; ++i4) {
    float4 v = xj4[i4];
    float a = al[(i4 * 4) >> 5];                 // head = channel/32
    atomicAdd(&o[i4 * 4 + 0], a * v.x);
    atomicAdd(&o[i4 * 4 + 1], a * v.y);
    atomicAdd(&o[i4 * 4 + 2], a * v.z);
    atomicAdd(&o[i4 * 4 + 3], a * v.w);
  }
}

// layer3: out = mean over heads + b3   (N x 32)
__global__ __launch_bounds__(256) void finalize_mean(const float* __restrict__ acc,
                                                     const float* __restrict__ b,
                                                     float* __restrict__ out, int count) {
  int i = blockIdx.x * blockDim.x + threadIdx.x;
  if (i >= count) return;
  int n = i / COUT, c = i % COUT;
  const float* a = acc + (size_t)n * HCDIM;
  out[i] = 0.25f * (a[c] + a[COUT + c] + a[2 * COUT + c] + a[3 * COUT + c]) + b[c];
}

// ---------------------------------------------------------------------------
// host side
// ---------------------------------------------------------------------------
static inline int cdiv(int a, int b) { return (a + b - 1) / b; }

static void run_layer(hipStream_t stream, int nN, int nE,
                      const bf16_t* xbs, const float* w, const float* q, const float* k,
                      const float* em, const float* le,
                      const int* ei, const int* etype, const float* eattr,
                      bf16_t* wbs, float* lee, float* xw, float* abuf,
                      unsigned* amax, float* denom, float* acc) {
  swizzle_w<<<cdiv(R_REL * 8 * 4 * 32, 256), 256, 0, stream>>>(w, wbs);
  compute_lee<<<1, 64, 0, stream>>>(le, em, lee);
  fill_u32<<<cdiv(nN * NHEAD, 256), 256, 0, stream>>>(amax, ENC_NEG_INF, nN * NHEAD);
  fill_f32<<<cdiv(nN * NHEAD, 256), 256, 0, stream>>>(denom, 0.f, nN * NHEAD);
  fill_f32<<<cdiv(nN * HCDIM, 256), 256, 0, stream>>>(acc, 0.f, nN * HCDIM);

  dim3 gg(cdiv(nN, 128), R_REL);
  gemm_xw_wmma<<<gg, 256, 0, stream>>>(xbs, wbs, xw, nN);

  int eb = cdiv(nE, 256);
  edge_alpha<<<eb, 256, 0, stream>>>(xw, ei, etype, eattr, q, k, lee, abuf, amax, nE, nN);
  edge_expsum<<<eb, 256, 0, stream>>>(abuf, amax, ei, denom, nE);
  edge_scatter<<<eb, 256, 0, stream>>>(abuf, denom, xw, ei, etype, acc, nE, nN);
}

extern "C" void kernel_launch(void* const* d_in, const int* in_sizes, int n_in,
                              void* d_out, int out_size, void* d_ws, size_t ws_size,
                              hipStream_t stream) {
  const float* x     = (const float*)d_in[0];
  const int*   ei    = (const int*)d_in[1];
  const float* eattr = (const float*)d_in[2];
  const int*   etype = (const int*)d_in[3];
  const float* w1 = (const float*)d_in[4];
  const float* q1 = (const float*)d_in[5];
  const float* k1 = (const float*)d_in[6];
  const float* e1 = (const float*)d_in[7];
  const float* le1 = (const float*)d_in[8];
  const float* b1 = (const float*)d_in[9];
  const float* w3 = (const float*)d_in[10];
  const float* q3 = (const float*)d_in[11];
  const float* k3 = (const float*)d_in[12];
  const float* e3 = (const float*)d_in[13];
  const float* le3 = (const float*)d_in[14];
  const float* b3 = (const float*)d_in[15];

  const int nN = in_sizes[0] / HCDIM;   // F_IN == H*C == 128
  const int nE = in_sizes[3];
  const int nTiles  = cdiv(nN, 128) * 8;      // 16-row tiles incl. padding
  const int xsTotal = nTiles * 4 * 32;        // fragments*lanes for swizzled X

  // workspace carve-out (256B-aligned)
  char* ws = (char*)d_ws;
  size_t off = 0;
  auto carve = [&](size_t bytes) {
    void* p = ws + off;
    off += (bytes + 255) & ~(size_t)255;
    return p;
  };
  float*    xw    = (float*)   carve((size_t)R_REL * nN * HCDIM * 4);  // 102.4 MB
  float*    abuf  = (float*)   carve((size_t)nE * NHEAD * 4);          // 12.8 MB
  unsigned* amax  = (unsigned*)carve((size_t)nN * NHEAD * 4);
  float*    denom = (float*)   carve((size_t)nN * NHEAD * 4);
  float*    acc   = (float*)   carve((size_t)nN * HCDIM * 4);          // 25.6 MB
  bf16_t*   xbs   = (bf16_t*)  carve((size_t)xsTotal * 16 * 2);        // swizzled X, ~12.8 MB
  bf16_t*   wbs   = (bf16_t*)  carve((size_t)R_REL * HCDIM * HCDIM * 2);
  float*    lee   = (float*)   carve((size_t)FE * NHEAD * 4);
  (void)ws_size;

  // ---- layer 1 (concat=True) ----
  swizzle_x<<<cdiv(xsTotal, 256), 256, 0, stream>>>(x, xbs, nN, xsTotal);
  run_layer(stream, nN, nE, xbs, w1, q1, k1, e1, le1, ei, etype, eattr,
            wbs, lee, xw, abuf, amax, denom, acc);
  // h = relu(acc + b1), fused into swizzled bf16 GEMM input of layer 3
  relu_bias_swizzle<<<cdiv(xsTotal, 256), 256, 0, stream>>>(acc, b1, xbs, nN, xsTotal);

  // ---- layer 3 (concat=False) ----
  run_layer(stream, nN, nE, xbs, w3, q3, k3, e3, le3, ei, etype, eattr,
            wbs, lee, xw, abuf, amax, denom, acc);
  finalize_mean<<<cdiv(nN * COUT, 256), 256, 0, stream>>>(acc, b3, (float*)d_out, nN * COUT);
}
